// Fastformer_85899346325
// MI455X (gfx1250) — compile-verified, matrix operand order checked by reading, not run on hardware
//
#include <hip/hip_runtime.h>
#include <hip/hip_bf16.h>

// ---------------- problem constants ----------------
constexpr int BB   = 4;
constexpr int LL   = 4096;
constexpr int HH   = 12;
constexpr int DHD  = 64;
constexpr int DD   = 768;            // = DIN = D
constexpr int MM   = BB * LL;        // 16384 rows
constexpr size_t SEQ = (size_t)BB * LL * DD;   // 12,582,912
constexpr size_t WEL = (size_t)DD * DD;        // 589,824
constexpr size_t ATT = (size_t)BB * HH * LL;   // 196,608
constexpr size_t GV  = (size_t)BB * HH * DHD;  // 3,072

typedef __attribute__((ext_vector_type(16))) __bf16 v16bf;
typedef __attribute__((ext_vector_type(8)))  __bf16 v8bf;
typedef __attribute__((ext_vector_type(8)))  float  v8f;

// ---------------- helpers ----------------
__device__ __forceinline__ unsigned short f32_to_bf16(float f) {
    unsigned int u = __float_as_uint(f);
    unsigned int r = u + 0x7FFFu + ((u >> 16) & 1u);   // RNE
    return (unsigned short)(r >> 16);
}

// Load a 16x32 bf16 fragment (A-matrix layout per CDNA5 ISA):
// lane m = lane&15 selects the row, half = lane>>4 selects K sub-ranges
// [h*8, h*8+8) into elems 0..7 and [16+h*8, ...) into elems 8..15.
__device__ __forceinline__ v16bf load_frag(const unsigned short* __restrict__ p) {
    v8bf lo = *(const v8bf*)(const void*)p;
    v8bf hi = *(const v8bf*)(const void*)(p + 16);
    v16bf r;
#pragma unroll
    for (int i = 0; i < 8; ++i) { r[i] = lo[i]; r[i + 8] = hi[i]; }
    return r;
}

// ---------------- f32 -> bf16 converters ----------------
__global__ __launch_bounds__(256) void cvt_f32_bf16_v4(
    const float* __restrict__ in, unsigned short* __restrict__ out) {
    size_t i4 = ((size_t)blockIdx.x * 256 + threadIdx.x) * 4;
    float4 v = *(const float4*)(in + i4);
    ushort4 o;
    o.x = f32_to_bf16(v.x); o.y = f32_to_bf16(v.y);
    o.z = f32_to_bf16(v.z); o.w = f32_to_bf16(v.w);
    *(ushort4*)(out + i4) = o;
}

// Transpose-convert weight [K,N] f32 -> [N,K] bf16  (768x768)
__global__ __launch_bounds__(256) void cvt_wT_bf16(
    const float* __restrict__ W, unsigned short* __restrict__ Wt) {
    int i = blockIdx.x * 256 + threadIdx.x;       // i = n*768 + k
    int n = i / DD, k = i - n * DD;
    Wt[i] = f32_to_bf16(W[(size_t)k * DD + n]);
}

// ---------------- bf16 WMMA GEMM:  D = A[M,K] * Bt[N,K]^T (+ Cadd) ----------------
__global__ __launch_bounds__(256) void gemm_bf16_wmma(
    const unsigned short* __restrict__ A,     // [M,K] bf16
    const unsigned short* __restrict__ Bt,    // [N,K] bf16 (pre-transposed)
    const float* __restrict__ Cadd,           // [M,N] f32 or nullptr
    float* __restrict__ Dst, int N, int K) {
    const int lane = threadIdx.x & 31;
    const int wid  = threadIdx.x >> 5;
    const int wm = wid >> 1, wn = wid & 1;            // 4x2 wave grid
    const int row0 = blockIdx.x * 128 + wm * 32;      // 2 m-tiles per wave
    const int col0 = blockIdx.y * 128 + wn * 64;      // 4 n-tiles per wave
    const int m  = lane & 15;
    const int hf = lane >> 4;
    const int mh = hf * 8;                            // accumulator row offset

    v8f acc[2][4];
    if (Cadd) {
#pragma unroll
        for (int i = 0; i < 2; ++i)
#pragma unroll
            for (int j = 0; j < 4; ++j)
#pragma unroll
                for (int r = 0; r < 8; ++r)
                    acc[i][j][r] = Cadd[(size_t)(row0 + i * 16 + mh + r) * N + col0 + j * 16 + m];
    } else {
#pragma unroll
        for (int i = 0; i < 2; ++i)
#pragma unroll
            for (int j = 0; j < 4; ++j)
#pragma unroll
                for (int r = 0; r < 8; ++r)
                    acc[i][j][r] = 0.0f;
    }

    const unsigned short* pa = A  + (size_t)(row0 + m) * K + hf * 8;
    const unsigned short* pb = Bt + (size_t)(col0 + m) * K + hf * 8;

    for (int kk = 0; kk < K; kk += 32) {
        __builtin_prefetch(pa + 64, 0, 0);   // global_prefetch A two K-blocks ahead
        __builtin_prefetch(pb + 64, 0, 0);
        v16bf a0 = load_frag(pa);
        v16bf a1 = load_frag(pa + (size_t)16 * K);
        v16bf b0 = load_frag(pb);
        v16bf b1 = load_frag(pb + (size_t)16 * K);
        v16bf b2 = load_frag(pb + (size_t)32 * K);
        v16bf b3 = load_frag(pb + (size_t)48 * K);

        acc[0][0] = __builtin_amdgcn_wmma_f32_16x16x32_bf16(false, a0, false, b0, (short)0, acc[0][0], false, false);
        acc[0][1] = __builtin_amdgcn_wmma_f32_16x16x32_bf16(false, a0, false, b1, (short)0, acc[0][1], false, false);
        acc[0][2] = __builtin_amdgcn_wmma_f32_16x16x32_bf16(false, a0, false, b2, (short)0, acc[0][2], false, false);
        acc[0][3] = __builtin_amdgcn_wmma_f32_16x16x32_bf16(false, a0, false, b3, (short)0, acc[0][3], false, false);
        acc[1][0] = __builtin_amdgcn_wmma_f32_16x16x32_bf16(false, a1, false, b0, (short)0, acc[1][0], false, false);
        acc[1][1] = __builtin_amdgcn_wmma_f32_16x16x32_bf16(false, a1, false, b1, (short)0, acc[1][1], false, false);
        acc[1][2] = __builtin_amdgcn_wmma_f32_16x16x32_bf16(false, a1, false, b2, (short)0, acc[1][2], false, false);
        acc[1][3] = __builtin_amdgcn_wmma_f32_16x16x32_bf16(false, a1, false, b3, (short)0, acc[1][3], false, false);
        pa += 32;
        pb += 32;
    }

#pragma unroll
    for (int i = 0; i < 2; ++i)
#pragma unroll
        for (int j = 0; j < 4; ++j)
#pragma unroll
            for (int r = 0; r < 8; ++r)
                Dst[(size_t)(row0 + i * 16 + mh + r) * N + col0 + j * 16 + m] = acc[i][j][r];
}

// ---------------- additive-attention logits: out[b,h,l] = scale * dot(X[b,l,:], W[:,h]) ----------------
__global__ __launch_bounds__(256) void logits12(
    const float* __restrict__ X,      // [B*L, 768]
    const float* __restrict__ Wsm,    // [768, 12]
    float* __restrict__ out,          // [B,H,L]
    float scale) {
    __shared__ float w[DD * HH];      // 36 KB in LDS
    for (int i = threadIdx.x; i < DD * HH; i += 256) w[i] = Wsm[i];
    __syncthreads();

    int rowid = blockIdx.x * 256 + threadIdx.x;     // 0..16383  (b,l)
    int b = rowid >> 12;
    int l = rowid & (LL - 1);
    const float* x = X + (size_t)rowid * DD;

    float acc[HH];
#pragma unroll
    for (int h = 0; h < HH; ++h) acc[h] = 0.0f;
    for (int d = 0; d < DD; ++d) {
        float xv = x[d];
#pragma unroll
        for (int h = 0; h < HH; ++h) acc[h] = fmaf(xv, w[d * HH + h], acc[h]);
    }
#pragma unroll
    for (int h = 0; h < HH; ++h)
        out[((size_t)(b * HH + h)) * LL + l] = acc[h] * scale;
}

// ---------------- in-place row softmax over L=4096 (one block per (b,h)) ----------------
__global__ __launch_bounds__(256) void softmax_rows(float* __restrict__ att) {
    float* row = att + (size_t)blockIdx.x * LL;
    int t = threadIdx.x;
    __shared__ float red[8];

    float mx = -3.4e38f;
    for (int i = t; i < LL; i += 256) mx = fmaxf(mx, row[i]);
#pragma unroll
    for (int o = 16; o; o >>= 1) mx = fmaxf(mx, __shfl_xor(mx, o, 32));
    if ((t & 31) == 0) red[t >> 5] = mx;
    __syncthreads();
    mx = red[0];
#pragma unroll
    for (int i = 1; i < 8; ++i) mx = fmaxf(mx, red[i]);
    __syncthreads();

    float sum = 0.0f;
    for (int i = t; i < LL; i += 256) {
        float e = __expf(row[i] - mx);
        row[i] = e;
        sum += e;
    }
#pragma unroll
    for (int o = 16; o; o >>= 1) sum += __shfl_xor(sum, o, 32);
    if ((t & 31) == 0) red[t >> 5] = sum;
    __syncthreads();
    sum = 0.0f;
#pragma unroll
    for (int i = 0; i < 8; ++i) sum += red[i];
    float inv = 1.0f / sum;
    for (int i = t; i < LL; i += 256) row[i] *= inv;
}

// ---------------- weighted pooling: out[b,h,d] = sum_l att[b,h,l] * X[off(b,h,l,d)] ----------------
__global__ __launch_bounds__(64) void pool64(
    const float* __restrict__ att, const float* __restrict__ X,
    float* __restrict__ out, long long sB, long long sH, long long sL) {
    int bh = blockIdx.x;
    int b = bh / HH, h = bh - b * HH;
    const float* a  = att + (size_t)bh * LL;
    const float* xp = X + (size_t)b * sB + (size_t)h * sH + threadIdx.x;
    float acc = 0.0f;
    for (int l = 0; l < LL; ++l) acc = fmaf(a[l], xp[(size_t)l * sL], acc);
    out[(size_t)bh * DHD + threadIdx.x] = acc;
}

// ---------------- qk[b,h,l,d] = Kp[b,l,h*64+d] * gq[b,h,d]  (qk stored contiguous [B,H,L,64]) ----------------
__global__ __launch_bounds__(256) void qk_mul(
    const float* __restrict__ Kp, const float* __restrict__ gq,
    float* __restrict__ qk) {
    size_t i4 = ((size_t)blockIdx.x * 256 + threadIdx.x) * 4;   // index in qk layout
    int d4 = (int)(i4 & 63);
    size_t rem = i4 >> 6;                // (b*H+h)*L + l
    int l  = (int)(rem & (LL - 1));
    int bh = (int)(rem >> 12);
    int b = bh / HH, h = bh - b * HH;
    float4 kv = *(const float4*)(Kp + ((size_t)(b * LL + l)) * DD + h * DHD + d4);
    float4 gv = *(const float4*)(gq + (size_t)bh * DHD + d4);
    float4 r;
    r.x = kv.x * gv.x; r.y = kv.y * gv.y; r.z = kv.z * gv.z; r.w = kv.w * gv.w;
    *(float4*)(qk + i4) = r;
}

// ---------------- P[b,l,c] = bf16( gk[b, c/64, c%64] * V[b,l,c] ) ----------------
__global__ __launch_bounds__(256) void p_mul_bf16(
    const float* __restrict__ V, const float* __restrict__ gk,
    unsigned short* __restrict__ P) {
    size_t i4 = ((size_t)blockIdx.x * 256 + threadIdx.x) * 4;   // [B,L,768] layout
    int c = (int)(i4 % DD);
    size_t row = i4 / DD;                 // b*L + l
    int b = (int)(row >> 12);
    int h = c >> 6, d = c & 63;
    float4 v = *(const float4*)(V + i4);
    float4 g = *(const float4*)(gk + (size_t)(b * HH + h) * DHD + d);
    ushort4 o;
    o.x = f32_to_bf16(v.x * g.x); o.y = f32_to_bf16(v.y * g.y);
    o.z = f32_to_bf16(v.z * g.z); o.w = f32_to_bf16(v.w * g.w);
    *(ushort4*)(P + i4) = o;
}

// ---------------- host orchestration ----------------
extern "C" void kernel_launch(void* const* d_in, const int* in_sizes, int n_in,
                              void* d_out, int out_size, void* d_ws, size_t ws_size,
                              hipStream_t stream) {
    const float* Qseq = (const float*)d_in[0];
    const float* Kseq = (const float*)d_in[1];
    const float* Vseq = (const float*)d_in[2];
    const float* WQ   = (const float*)d_in[3];
    const float* WK   = (const float*)d_in[4];
    const float* WV   = (const float*)d_in[5];
    const float* Wq   = (const float*)d_in[6];
    const float* Wk   = (const float*)d_in[7];
    const float* WP   = (const float*)d_in[8];
    float* out = (float*)d_out;

    // carve workspace (256B aligned regions)
    char* base = (char*)d_ws;
    size_t off = 0;
    auto carve = [&](size_t bytes) -> char* {
        char* p = base + off;
        off = (off + bytes + 255) & ~(size_t)255;
        return p;
    };
    unsigned short* Qbf  = (unsigned short*)carve(SEQ * 2);
    unsigned short* Kbf  = (unsigned short*)carve(SEQ * 2);
    unsigned short* Vbf  = (unsigned short*)carve(SEQ * 2);
    unsigned short* WQt  = (unsigned short*)carve(WEL * 2);
    unsigned short* WKt  = (unsigned short*)carve(WEL * 2);
    unsigned short* WVt  = (unsigned short*)carve(WEL * 2);
    unsigned short* WPt  = (unsigned short*)carve(WEL * 2);
    float* Qf  = (float*)carve(SEQ * 4);
    float* Kpf = (float*)carve(SEQ * 4);
    float* Vf  = (float*)carve(SEQ * 4);
    float* qk  = (float*)carve(SEQ * 4);
    float* att1 = (float*)carve(ATT * 4);
    float* att2 = (float*)carve(ATT * 4);
    float* gq  = (float*)carve(GV * 4);
    float* gk  = (float*)carve(GV * 4);
    unsigned short* Pbf = Qbf;   // reuse: Qbf dead after first GEMM

    const float scale = 0.125f;  // 1/sqrt(64)
    const dim3 gemmGrid(MM / 128, DD / 128);

    // 1) convert sequences + weights to bf16 (weights transposed to [N,K])
    cvt_f32_bf16_v4<<<(unsigned)(SEQ / 1024), 256, 0, stream>>>(Qseq, Qbf);
    cvt_f32_bf16_v4<<<(unsigned)(SEQ / 1024), 256, 0, stream>>>(Kseq, Kbf);
    cvt_f32_bf16_v4<<<(unsigned)(SEQ / 1024), 256, 0, stream>>>(Vseq, Vbf);
    cvt_wT_bf16<<<(unsigned)(WEL / 256), 256, 0, stream>>>(WQ, WQt);
    cvt_wT_bf16<<<(unsigned)(WEL / 256), 256, 0, stream>>>(WK, WKt);
    cvt_wT_bf16<<<(unsigned)(WEL / 256), 256, 0, stream>>>(WV, WVt);
    cvt_wT_bf16<<<(unsigned)(WEL / 256), 256, 0, stream>>>(WP, WPt);

    // 2) projections (bf16 WMMA, f32 accumulate)
    gemm_bf16_wmma<<<gemmGrid, 256, 0, stream>>>(Qbf, WQt, nullptr, Qf,  DD, DD);
    gemm_bf16_wmma<<<gemmGrid, 256, 0, stream>>>(Kbf, WKt, nullptr, Kpf, DD, DD);
    gemm_bf16_wmma<<<gemmGrid, 256, 0, stream>>>(Vbf, WVt, nullptr, Vf,  DD, DD);

    // 3) query attention -> global_q
    logits12<<<MM / 256, 256, 0, stream>>>(Qf, Wq, att1, scale);
    softmax_rows<<<BB * HH, 256, 0, stream>>>(att1);
    pool64<<<BB * HH, 64, 0, stream>>>(att1, Qf, gq,
                                       (long long)LL * DD, (long long)DHD, (long long)DD);

    // 4) qk = Kh * global_q  (stored [B,H,L,64]; same memory IS qk_flat [B,L,768])
    qk_mul<<<(unsigned)(SEQ / 1024), 256, 0, stream>>>(Kpf, gq, qk);

    // 5) key attention -> global_k
    logits12<<<MM / 256, 256, 0, stream>>>(qk, Wk, att2, scale);
    softmax_rows<<<BB * HH, 256, 0, stream>>>(att2);
    pool64<<<BB * HH, 64, 0, stream>>>(att2, qk, gk,
                                       (long long)HH * LL * DHD, (long long)LL * DHD,
                                       (long long)DHD);

    // 6) P = global_k * Vh  (already in [B,L,768] order), then out = P@WP + Q (residual fused as C)
    p_mul_bf16<<<(unsigned)(SEQ / 1024), 256, 0, stream>>>(Vf, gk, Pbf);
    gemm_bf16_wmma<<<gemmGrid, 256, 0, stream>>>(Pbf, WPt, Qf, out, DD, DD);

    (void)in_sizes; (void)n_in; (void)out_size; (void)ws_size;
}
